// TripletTransformerNetwork_31009663877393
// MI455X (gfx1250) — compile-verified, matrix operand order checked by reading, not run on hardware
//
#include <hip/hip_runtime.h>
#include <hip/hip_bf16.h>
#include <math.h>

#define NODES 20000
#define EDGES 320000
#define DIN   128
#define NCLS  10

typedef __attribute__((ext_vector_type(16))) __bf16 v16bf;
typedef __attribute__((ext_vector_type(8)))  __bf16 bf16x8;
typedef __attribute__((ext_vector_type(8)))  float  v8f;

// ---- order-preserving float<->uint encoding for atomicMax on floats ----
__device__ __forceinline__ unsigned fenc(float f) {
    unsigned u = __float_as_uint(f);
    return (u & 0x80000000u) ? ~u : (u | 0x80000000u);
}
__device__ __forceinline__ float fdec(unsigned k) {
    return __uint_as_float((k & 0x80000000u) ? (k & 0x7FFFFFFFu) : ~k);
}
__device__ __forceinline__ float gelu_exact(float x) {
    return 0.5f * x * (1.0f + erff(x * 0.7071067811865475f));
}

// ---- fp32 -> bf16 row-major copy (activations, once per layer) ----
__global__ void k_xcvt(const float* __restrict__ X, __bf16* __restrict__ Xb, int total) {
    int i = blockIdx.x * blockDim.x + threadIdx.x;
    if (i >= total) return;
    Xb[i] = (__bf16)X[i];
}

// ---- transpose weight [di,do] -> bf16 [do,di] so B-fragments load contiguously
__global__ void k_wt(const float* __restrict__ W, __bf16* __restrict__ WT, int di, int dout) {
    int i = blockIdx.x * blockDim.x + threadIdx.x;
    if (i >= di * dout) return;
    int c = i / di, k = i % di;
    WT[i] = (__bf16)W[(size_t)k * dout + c];
}

// ---- WMMA GEMM: Y[n,dout] = X[n,di] @ W + bias
// Xb: bf16 [n,di] row-major. WT: bf16 [dout,di] (transposed weight).
// Block = 64 threads (2 waves). Each wave computes a 16x64 output strip:
// one A fragment per k-step feeds 4 WMMAs (4 N-tiles, accumulators in regs).
__global__ void k_gemm(const __bf16* __restrict__ Xb, const __bf16* __restrict__ WT,
                       const float* __restrict__ bias, float* __restrict__ Y,
                       int di, int dout) {
    int lane = threadIdx.x & 31;
    int wave = threadIdx.x >> 5;            // 0..1
    int half = lane >> 4;                   // K-halves per WMMA bf16 layout
    int l15  = lane & 15;
    int m0 = (blockIdx.x * 2 + wave) * 16;  // 1250 M-tiles / 2 waves = 625 blocks
    int c0 = blockIdx.y * 64;               // 4 N-tiles per wave

    const __bf16* xr = Xb + (size_t)(m0 + l15) * di + 8 * half;
    const __bf16* w0 = WT + (size_t)(c0 +      l15) * di + 8 * half;
    const __bf16* w1 = w0 + (size_t)16 * di;
    const __bf16* w2 = w0 + (size_t)32 * di;
    const __bf16* w3 = w0 + (size_t)48 * di;

    v8f acc0 = {}, acc1 = {}, acc2 = {}, acc3 = {};
    for (int k0 = 0; k0 < di; k0 += 32) {
        bf16x8 alo = *(const bf16x8*)(xr + k0);
        bf16x8 ahi = *(const bf16x8*)(xr + k0 + 16);
        v16bf a = __builtin_shufflevector(alo, ahi,
                   0,1,2,3,4,5,6,7,8,9,10,11,12,13,14,15);
        bf16x8 b0l = *(const bf16x8*)(w0 + k0), b0h = *(const bf16x8*)(w0 + k0 + 16);
        bf16x8 b1l = *(const bf16x8*)(w1 + k0), b1h = *(const bf16x8*)(w1 + k0 + 16);
        bf16x8 b2l = *(const bf16x8*)(w2 + k0), b2h = *(const bf16x8*)(w2 + k0 + 16);
        bf16x8 b3l = *(const bf16x8*)(w3 + k0), b3h = *(const bf16x8*)(w3 + k0 + 16);
        v16bf b0 = __builtin_shufflevector(b0l, b0h, 0,1,2,3,4,5,6,7,8,9,10,11,12,13,14,15);
        v16bf b1 = __builtin_shufflevector(b1l, b1h, 0,1,2,3,4,5,6,7,8,9,10,11,12,13,14,15);
        v16bf b2 = __builtin_shufflevector(b2l, b2h, 0,1,2,3,4,5,6,7,8,9,10,11,12,13,14,15);
        v16bf b3 = __builtin_shufflevector(b3l, b3h, 0,1,2,3,4,5,6,7,8,9,10,11,12,13,14,15);
        acc0 = __builtin_amdgcn_wmma_f32_16x16x32_bf16(false, a, false, b0, (short)0, acc0, false, false);
        acc1 = __builtin_amdgcn_wmma_f32_16x16x32_bf16(false, a, false, b1, (short)0, acc1, false, false);
        acc2 = __builtin_amdgcn_wmma_f32_16x16x32_bf16(false, a, false, b2, (short)0, acc2, false, false);
        acc3 = __builtin_amdgcn_wmma_f32_16x16x32_bf16(false, a, false, b3, (short)0, acc3, false, false);
    }
    float bv0 = bias[c0 + l15];
    float bv1 = bias[c0 + 16 + l15];
    float bv2 = bias[c0 + 32 + l15];
    float bv3 = bias[c0 + 48 + l15];
#pragma unroll
    for (int r = 0; r < 8; ++r) {
        size_t mrow = (size_t)(m0 + r + 8 * half) * dout;
        Y[mrow + c0      + l15] = acc0[r] + bv0;
        Y[mrow + c0 + 16 + l15] = acc1[r] + bv1;
        Y[mrow + c0 + 32 + l15] = acc2[r] + bv2;
        Y[mrow + c0 + 48 + l15] = acc3[r] + bv3;
    }
}

// ---- init per-(node,head) running max (encoded) and denom
__global__ void k_init_md(unsigned* __restrict__ menc, float* __restrict__ den, int total) {
    int i = blockIdx.x * blockDim.x + threadIdx.x;
    if (i >= total) return;
    menc[i] = fenc(-INFINITY);
    den[i]  = 0.0f;
}

// ---- pass 1: logits + segment max over dst
__global__ void k_edge_logits(const int* __restrict__ src, const int* __restrict__ dst,
                              const float* __restrict__ Q, const float* __restrict__ K,
                              float* __restrict__ logits, unsigned* __restrict__ menc,
                              int Hn, int C) {
    int i = blockIdx.x * blockDim.x + threadIdx.x;
    if (i >= EDGES * Hn) return;
    int e = i / Hn, h = i % Hn;
    int s = src[e], d = dst[e];
    const float* qp = Q + (size_t)d * Hn * C + h * C;
    const float* kp = K + (size_t)s * Hn * C + h * C;
    float acc = 0.0f;
    for (int c = 0; c < C; ++c) acc += qp[c] * kp[c];
    acc /= sqrtf((float)C);
    logits[i] = acc;
    atomicMax(&menc[(size_t)d * Hn + h], fenc(acc));
}

// ---- pass 2: e = exp(logit - max) ; segment sum over dst
__global__ void k_edge_exp(const int* __restrict__ dst, float* __restrict__ logits,
                           const unsigned* __restrict__ menc, float* __restrict__ den, int Hn) {
    int i = blockIdx.x * blockDim.x + threadIdx.x;
    if (i >= EDGES * Hn) return;
    int e = i / Hn, h = i % Hn;
    int d = dst[e];
    float m  = fdec(menc[(size_t)d * Hn + h]);
    float ex = expf(logits[i] - m);
    logits[i] = ex;
    atomicAdd(&den[(size_t)d * Hn + h], ex);
}

// ---- pass 3: alpha = e/denom ; agg[dst] += alpha * v[src] (agg pre-seeded with skip)
__global__ void k_edge_agg(const int* __restrict__ src, const int* __restrict__ dst,
                           const float* __restrict__ logits, const float* __restrict__ den,
                           const float* __restrict__ V, float* __restrict__ agg,
                           float* __restrict__ alpha_out, int Hn, int C) {
    int i = blockIdx.x * blockDim.x + threadIdx.x;
    if (i >= EDGES * Hn) return;
    int e = i / Hn, h = i % Hn;
    int s = src[e], d = dst[e];
    float a = logits[i] / (den[(size_t)d * Hn + h] + 1e-16f);
    alpha_out[i] = a;
    const float* vp = V   + (size_t)s * Hn * C + h * C;
    float*       op = agg + (size_t)d * Hn * C + h * C;
    for (int c = 0; c < C; ++c) atomicAdd(&op[c], a * vp[c]);
}

// ---- GeLU(exact) + LayerNorm, one block per node, d = power of two (64..256)
__global__ void k_postln(const float* __restrict__ in, float* __restrict__ out,
                         const float* __restrict__ g, const float* __restrict__ b, int d) {
    extern __shared__ float sm[];
    int n = blockIdx.x, t = threadIdx.x;
    float x = gelu_exact(in[(size_t)n * d + t]);
    sm[t] = x; __syncthreads();
    for (int s = d >> 1; s > 0; s >>= 1) { if (t < s) sm[t] += sm[t + s]; __syncthreads(); }
    float mu = sm[0] / d; __syncthreads();
    float dx = x - mu;
    sm[t] = dx * dx; __syncthreads();
    for (int s = d >> 1; s > 0; s >>= 1) { if (t < s) sm[t] += sm[t + s]; __syncthreads(); }
    float var = sm[0] / d;
    out[(size_t)n * d + t] = g[t] * dx * rsqrtf(var + 1e-5f) + b[t];
}

// ---- classifier: x_out = h2 @ fc_W + fc_b  ([N,64] @ [64,10])
__global__ void k_fc(const float* __restrict__ h2, const float* __restrict__ W,
                     const float* __restrict__ b, float* __restrict__ out) {
    int i = blockIdx.x * blockDim.x + threadIdx.x;
    if (i >= NODES * NCLS) return;
    int n = i / NCLS, c = i % NCLS;
    float acc = b[c];
#pragma unroll
    for (int k = 0; k < 64; ++k) acc += h2[(size_t)n * 64 + k] * W[k * NCLS + c];
    out[i] = acc;
}

// ================= host-side layer driver =================
static void run_layer(hipStream_t stream, const float* Xin, int di, int dout, int Hn,
                      const float* Wq, const float* Wk, const float* Wv, const float* Ws,
                      const float* bq, const float* bk, const float* bv, const float* bs,
                      const float* ng, const float* nb,
                      const int* src, const int* dst,
                      float* Q, float* K, float* V, float* AGG,
                      float* LOG, unsigned* MENC, float* DEN,
                      __bf16* WT, __bf16* Xb,
                      float* Yout, float* alpha_out) {
    int C = dout / Hn;
    __bf16* WTq = WT;
    __bf16* WTk = WT + 65536;
    __bf16* WTv = WT + 2 * 65536;
    __bf16* WTs = WT + 3 * 65536;
    int wtot = di * dout;
    int wblk = (wtot + 255) / 256;
    k_wt<<<wblk, 256, 0, stream>>>(Wq, WTq, di, dout);
    k_wt<<<wblk, 256, 0, stream>>>(Wk, WTk, di, dout);
    k_wt<<<wblk, 256, 0, stream>>>(Wv, WTv, di, dout);
    k_wt<<<wblk, 256, 0, stream>>>(Ws, WTs, di, dout);

    int xtot = NODES * di;
    k_xcvt<<<(xtot + 255) / 256, 256, 0, stream>>>(Xin, Xb, xtot);

    dim3 gg(NODES / 32, dout / 64);      // 625 x {1,2,4}
    k_gemm<<<gg, 64, 0, stream>>>(Xb, WTq, bq, Q,   di, dout);
    k_gemm<<<gg, 64, 0, stream>>>(Xb, WTk, bk, K,   di, dout);
    k_gemm<<<gg, 64, 0, stream>>>(Xb, WTv, bv, V,   di, dout);
    k_gemm<<<gg, 64, 0, stream>>>(Xb, WTs, bs, AGG, di, dout); // skip seeds agg

    int nh = NODES * Hn;
    k_init_md<<<(nh + 255) / 256, 256, 0, stream>>>(MENC, DEN, nh);

    int eh = EDGES * Hn;
    int eblk = (eh + 255) / 256;
    k_edge_logits<<<eblk, 256, 0, stream>>>(src, dst, Q, K, LOG, MENC, Hn, C);
    k_edge_exp   <<<eblk, 256, 0, stream>>>(dst, LOG, MENC, DEN, Hn);
    k_edge_agg   <<<eblk, 256, 0, stream>>>(src, dst, LOG, DEN, V, AGG,
                                            alpha_out ? alpha_out : LOG, Hn, C);

    k_postln<<<NODES, dout, dout * sizeof(float), stream>>>(AGG, Yout, ng, nb, dout);
}

extern "C" void kernel_launch(void* const* d_in, const int* in_sizes, int n_in,
                              void* d_out, int out_size, void* d_ws, size_t ws_size,
                              hipStream_t stream) {
    (void)in_sizes; (void)n_in; (void)out_size; (void)ws_size;
    const float* x  = (const float*)d_in[0];
    const int*   ei = (const int*)d_in[1];
    const int* src = ei;
    const int* dst = ei + EDGES;
    auto f = [&](int i) { return (const float*)d_in[i]; };

    // workspace layout (floats)
    float* ws = (float*)d_ws;
    size_t o = 0;
    float* h1  = ws + o; o += (size_t)NODES * 256;
    float* h2  = ws + o; o += (size_t)NODES * 64;
    float* h3  = ws + o; o += (size_t)NODES * 256;
    float* Q   = ws + o; o += (size_t)NODES * 256;
    float* K   = ws + o; o += (size_t)NODES * 256;
    float* V   = ws + o; o += (size_t)NODES * 256;
    float* AGG = ws + o; o += (size_t)NODES * 256;
    float* LOG = ws + o; o += (size_t)EDGES * 8;
    unsigned* MENC = (unsigned*)(ws + o); o += (size_t)NODES * 8;
    float* DEN = ws + o; o += (size_t)NODES * 8;
    __bf16* WT = (__bf16*)(ws + o); o += (size_t)4 * 256 * 256 / 2;  // 4 x 256x256 bf16
    __bf16* Xb = (__bf16*)(ws + o); o += (size_t)NODES * 256 / 2;    // bf16 activations

    // output packing: x_out [N,10] | h4 [N,128] | attn [E,8]
    float* out_x    = (float*)d_out;
    float* out_h4   = out_x  + (size_t)NODES * NCLS;
    float* out_attn = out_h4 + (size_t)NODES * DIN;

    // layer 1: 128 -> 256, H=8, C=32
    run_layer(stream, x,  128, 256, 8, f(2), f(3), f(4), f(5), f(6), f(7), f(8), f(9),
              f(34), f(35), src, dst, Q, K, V, AGG, LOG, MENC, DEN, WT, Xb, h1, nullptr);
    // layer 2: 256 -> 64, H=8, C=8 ; alpha -> d_out attn region
    run_layer(stream, h1, 256, 64, 8, f(10), f(11), f(12), f(13), f(14), f(15), f(16), f(17),
              f(36), f(37), src, dst, Q, K, V, AGG, LOG, MENC, DEN, WT, Xb, h2, out_attn);
    // layer 3: 64 -> 256, H=8, C=32
    run_layer(stream, h2, 64, 256, 8, f(18), f(19), f(20), f(21), f(22), f(23), f(24), f(25),
              f(38), f(39), src, dst, Q, K, V, AGG, LOG, MENC, DEN, WT, Xb, h3, nullptr);
    // layer 4: 256 -> 128, H=1, C=128 ; output straight into h4 region
    run_layer(stream, h3, 256, 128, 1, f(26), f(27), f(28), f(29), f(30), f(31), f(32), f(33),
              f(40), f(41), src, dst, Q, K, V, AGG, LOG, MENC, DEN, WT, Xb, out_h4, nullptr);

    // classifier head on h2
    k_fc<<<(NODES * NCLS + 255) / 256, 256, 0, stream>>>(h2, f(42), f(43), out_x);
}
